// SparseNGCNLayer_13357348290974
// MI455X (gfx1250) — compile-verified, matrix operand order checked by reading, not run on hardware
//
#include <hip/hip_runtime.h>

#define IN_CH  512
#define OUT_CH 128

typedef __attribute__((ext_vector_type(16))) __bf16 v16bf;
typedef __attribute__((ext_vector_type(8)))  float  v8f;

// ---------------------------------------------------------------------------
// 1) Densify sparse features: dense[row][col] += val  (duplicates sum, like
//    segment_sum in the reference).
// ---------------------------------------------------------------------------
__global__ void scatter_feat_kernel(const int* __restrict__ rows,
                                    const int* __restrict__ cols,
                                    const float* __restrict__ vals,
                                    float* __restrict__ dense, int nnz) {
  int i = blockIdx.x * blockDim.x + threadIdx.x;
  if (i < nnz) {
    atomicAdd(&dense[(size_t)rows[i] * IN_CH + cols[i]], vals[i]);
  }
}

// ---------------------------------------------------------------------------
// 2) Convert weight f32 [512x128] -> bf16 transposed [128x512] so B-matrix
//    WMMA fragments are contiguous along K.
// ---------------------------------------------------------------------------
__global__ void weight_to_bf16t_kernel(const float* __restrict__ W,
                                       __bf16* __restrict__ Wt, int total) {
  int i = blockIdx.x * blockDim.x + threadIdx.x;
  if (i < total) {
    int k = i / OUT_CH;
    int n = i - k * OUT_CH;
    Wt[(size_t)n * IN_CH + k] = (__bf16)W[i];
  }
}

// ---------------------------------------------------------------------------
// 3) Dense GEMM via v_wmma_f32_16x16x32_bf16, fused bias + ReLU.
//    One wave computes a 16(M) x 128(N) strip: 8 N-tiles of 16x16 f32 accum.
//    Fragment layouts per CDNA5 ISA 7.12.2:
//      A (16x32 bf16): lane&15 = M row; lane half selects K {0-7,16-23} vs
//                      {8-15,24-31}; reg r<4 -> K=2r+8*half, r>=4 -> +16.
//      B (32x16 bf16): same K mapping, lane&15 = N column (Wt is KxN^T).
//      C/D (16x16 f32): VGPR j -> row = j + 8*half, col = lane&15.
// ---------------------------------------------------------------------------
__global__ void __launch_bounds__(256)
spmm_wmma_bias_relu_kernel(const float* __restrict__ A,     // [M, 512] f32
                           const __bf16* __restrict__ Wt,   // [128, 512] bf16
                           const float* __restrict__ bias,  // [128]
                           float* __restrict__ out,         // [M, 128]
                           int M) {
  const int waveInBlk = threadIdx.x >> 5;
  const int lane      = threadIdx.x & 31;
  const int mtile     = blockIdx.x * 8 + waveInBlk;
  const int m0        = mtile * 16;
  if (m0 >= M) return;                  // wave-uniform: EXEC stays all-ones

  const int lrow = lane & 15;
  const int half = lane >> 4;

  int arow_i = m0 + lrow;
  if (arow_i >= M) arow_i = M - 1;      // clamp (M=100000 is 16-aligned anyway)
  const float* arow = A + (size_t)arow_i * IN_CH;

  v8f acc[8] = {};

  for (int k0 = 0; k0 < IN_CH; k0 += 32) {
    // ---- A fragment: convert f32 -> bf16 while loading
    const float* ap = arow + k0 + 8 * half;
    v16bf a;
#pragma unroll
    for (int r = 0; r < 4; ++r) {
      a[2 * r]         = (__bf16)ap[2 * r];
      a[2 * r + 1]     = (__bf16)ap[2 * r + 1];
      a[8 + 2 * r]     = (__bf16)ap[16 + 2 * r];
      a[8 + 2 * r + 1] = (__bf16)ap[16 + 2 * r + 1];
    }
    // ---- 8 N-tiles of B, accumulate
#pragma unroll
    for (int nt = 0; nt < 8; ++nt) {
      const __bf16* bp = Wt + (size_t)(nt * 16 + lrow) * IN_CH + k0 + 8 * half;
      v16bf b;
#pragma unroll
      for (int r = 0; r < 4; ++r) {
        b[2 * r]         = bp[2 * r];
        b[2 * r + 1]     = bp[2 * r + 1];
        b[8 + 2 * r]     = bp[16 + 2 * r];
        b[8 + 2 * r + 1] = bp[16 + 2 * r + 1];
      }
      acc[nt] = __builtin_amdgcn_wmma_f32_16x16x32_bf16(
          false, a, false, b, (short)0, acc[nt], false, false);
    }
  }

  // ---- epilogue: bias + ReLU, scatter per C/D layout
#pragma unroll
  for (int nt = 0; nt < 8; ++nt) {
    const int col = nt * 16 + lrow;
    const float bv = bias[col];
#pragma unroll
    for (int j = 0; j < 8; ++j) {
      int row = m0 + j + 8 * half;
      if (row < M) {
        float v = acc[nt][j] + bv;
        out[(size_t)row * OUT_CH + col] = v > 0.f ? v : 0.f;
      }
    }
  }
}

// ---------------------------------------------------------------------------
// 4) Graph propagation: out[dst] += val * in[src], one wave per edge,
//    each lane handles 4 channels (float4 gather, 4 f32 atomics).
// ---------------------------------------------------------------------------
__global__ void prop_kernel(const int* __restrict__ dst_rows,
                            const int* __restrict__ src_cols,
                            const float* __restrict__ vals,
                            const float* __restrict__ in,
                            float* __restrict__ out, int nedges) {
  const int gw   = (blockIdx.x * blockDim.x + threadIdx.x) >> 5;
  const int lane = threadIdx.x & 31;
  if (gw >= nedges) return;
  const int dst = dst_rows[gw];
  const int src = src_cols[gw];
  const float v = vals[gw];
  const float4* ip = (const float4*)(in + (size_t)src * OUT_CH);
  float4 x = ip[lane];
  float* op = out + (size_t)dst * OUT_CH + lane * 4;
  atomicAdd(op + 0, v * x.x);
  atomicAdd(op + 1, v * x.y);
  atomicAdd(op + 2, v * x.z);
  atomicAdd(op + 3, v * x.w);
}

// ---------------------------------------------------------------------------
extern "C" void kernel_launch(void* const* d_in, const int* in_sizes, int n_in,
                              void* d_out, int out_size, void* d_ws, size_t ws_size,
                              hipStream_t stream) {
  const int*   feat_rows = (const int*)d_in[0];
  const int*   feat_cols = (const int*)d_in[1];
  const float* feat_vals = (const float*)d_in[2];
  const int*   adj_rows  = (const int*)d_in[3];
  const int*   adj_cols  = (const int*)d_in[4];
  const float* adj_vals  = (const float*)d_in[5];
  const float* weight    = (const float*)d_in[6];
  const float* bias      = (const float*)d_in[7];

  const int nnz    = in_sizes[0];
  const int nedges = in_sizes[3];
  const int out_ch = in_sizes[7];          // 128
  const int M      = out_size / out_ch;    // n_nodes (100000)

  // workspace layout: dense f32 features | ping buffer | bf16 W^T
  char* ws = (char*)d_ws;
  size_t off = 0;
  float* denseA = (float*)(ws + off);
  off += (((size_t)M * IN_CH * sizeof(float)) + 255) & ~(size_t)255;
  float* ping = (float*)(ws + off);
  off += (((size_t)M * OUT_CH * sizeof(float)) + 255) & ~(size_t)255;
  __bf16* Wt = (__bf16*)(ws + off);
  (void)ws_size; (void)n_in;

  // 1) densify features
  hipMemsetAsync(denseA, 0, (size_t)M * IN_CH * sizeof(float), stream);
  scatter_feat_kernel<<<(nnz + 255) / 256, 256, 0, stream>>>(
      feat_rows, feat_cols, feat_vals, denseA, nnz);

  // 2) weight -> bf16 transposed
  const int wtotal = in_sizes[6];
  weight_to_bf16t_kernel<<<(wtotal + 255) / 256, 256, 0, stream>>>(weight, Wt, wtotal);

  // 3) base = relu(dense @ W + bias) via WMMA, written to d_out
  const int mtiles = (M + 15) / 16;
  spmm_wmma_bias_relu_kernel<<<(mtiles + 7) / 8, 256, 0, stream>>>(
      denseA, Wt, bias, (float*)d_out, M);

  // 4) two propagation hops: d_out -> ping -> d_out
  hipMemsetAsync(ping, 0, (size_t)M * OUT_CH * sizeof(float), stream);
  prop_kernel<<<(nedges + 7) / 8, 256, 0, stream>>>(
      adj_rows, adj_cols, adj_vals, (const float*)d_out, ping, nedges);

  hipMemsetAsync(d_out, 0, (size_t)out_size * sizeof(float), stream);
  prop_kernel<<<(nedges + 7) / 8, 256, 0, stream>>>(
      adj_rows, adj_cols, adj_vals, ping, (float*)d_out, nedges);
}
